// MultiHeadAttention_26104811225156
// MI455X (gfx1250) — matrix operand order repaired and checked by reading, not compiled, not run on hardware
//
#include <hip/hip_runtime.h>

// ---------------------------------------------------------------------------
// MI455X (gfx1250) multi-head attention, bf16 WMMA pipeline.
//   B=2, S=2048, D=1024, H=16, HD=64
// Workspace layout (bf16), ~62 MB total:
//   Qb,Kb,Vb   [B,S,D]       8 MB each
//   Wqb,Wvb    [H,D,HD]      2 MB each
//   Wob        [D,D]         2 MB
//   qp,kp,vp   [B,H,S,HD]    8 MB each   (K projected with Wq per reference bug)
//   concat     [B,S,D]       8 MB
// ---------------------------------------------------------------------------

#define BB 2
#define SS 2048
#define DD 1024
#define HH 16
#define HD 64

typedef __bf16 bf16_t;
typedef __attribute__((ext_vector_type(16))) __bf16 v16bf;
typedef __attribute__((ext_vector_type(8)))  float  v8f;

struct alignas(16) Q16 { unsigned int a, b, c, d; };   // 16-byte POD mover
struct alignas(16) F4  { float x, y, z, w; };

union U16x16 { v16bf v; Q16 q[2]; bf16_t e[16]; };
union U16x8  { Q16 q;   bf16_t e[8]; };

// ---- async global->LDS path (ASYNCcnt) with synchronous fallback ----------
#if defined(__gfx1250__) && defined(__has_builtin)
#if __has_builtin(__builtin_amdgcn_global_load_async_to_lds_b128)
#define HAS_ASYNC_LDS 1
#endif
#endif
#ifndef HAS_ASYNC_LDS
#define HAS_ASYNC_LDS 0
#endif

#if HAS_ASYNC_LDS
typedef int v4i __attribute__((vector_size(16)));
typedef __attribute__((address_space(1))) v4i g_v4i;   // global int4
typedef __attribute__((address_space(3))) v4i l_v4i;   // LDS int4
#endif

// copy 16B from global to LDS (per-lane addresses)
static __device__ inline void copy16(bf16_t* lds, const bf16_t* g) {
#if HAS_ASYNC_LDS
  __attribute__((address_space(1))) void* gp =
      (__attribute__((address_space(1))) void*)(g);
  __attribute__((address_space(3))) void* lp =
      (__attribute__((address_space(3))) void*)(lds);
  __builtin_amdgcn_global_load_async_to_lds_b128(
      (g_v4i*)gp, (l_v4i*)lp, /*offset=*/0, /*cpol=*/0);
#else
  *reinterpret_cast<Q16*>(lds) = *reinterpret_cast<const Q16*>(g);
#endif
}

static __device__ inline void tiles_wait() {
#if HAS_ASYNC_LDS
#if __has_builtin(__builtin_amdgcn_s_wait_asynccnt)
  __builtin_amdgcn_s_wait_asynccnt(0);
#else
  asm volatile("s_wait_asynccnt 0x0" ::: "memory");
#endif
#endif
}

static __device__ inline v8f zero8() {
  v8f z = {0.f, 0.f, 0.f, 0.f, 0.f, 0.f, 0.f, 0.f};
  return z;
}

static __device__ inline v8f wmma_bf16(v16bf a, v16bf b, v8f c) {
  // D = A(16x32 bf16) * B(32x16 bf16) + C(16x16 f32)
  return __builtin_amdgcn_wmma_f32_16x16x32_bf16(
      /*neg_a=*/false, a, /*neg_b=*/false, b,
      /*c_mod=*/(short)0, c, /*reuse_a=*/false, /*reuse_b=*/false);
}

// A fragment, 16-bit 16x32 (MxK), row-major LDS tile, leading dim `ld` elems
// (ld multiple of 8 so 16B loads stay aligned).
// ISA layout: lanes 0-15 hold K {0..7,16..23}, lanes 16-31 K {8..15,24..31}.
static __device__ inline v16bf frag_a(const bf16_t* base, int ld) {
  const int lane = threadIdx.x & 31;
  const int row  = lane & 15;
  const int kb   = (lane >> 4) << 3;  // 0 or 8
  U16x16 u;
  u.q[0] = *reinterpret_cast<const Q16*>(base + row * ld + kb);
  u.q[1] = *reinterpret_cast<const Q16*>(base + row * ld + kb + 16);
  return u.v;
}

// B fragment, 16-bit 32x16 (KxN). `base` -> LDS tile stored BT[n][k] row-major
// (n = output column, k = reduction). Lane-group g holds K {g*8..g*8+7, 16+g*8..16+g*8+7}.
static __device__ inline v16bf frag_b(const bf16_t* base, int ld) {
  const int lane = threadIdx.x & 31;
  const int col  = lane & 15;
  const int kb   = (lane >> 4) << 3;  // 0 or 8
  U16x16 u;
  u.q[0] = *reinterpret_cast<const Q16*>(base + col * ld + kb);
  u.q[1] = *reinterpret_cast<const Q16*>(base + col * ld + kb + 16);
  return u.v;
}

// ---------------------------------------------------------------------------
// Kernel 0: fp32 -> bf16 bulk convert (n multiple of 2048)
// ---------------------------------------------------------------------------
__global__ __launch_bounds__(256) void convert_bf16_kernel(
    const float* __restrict__ src, bf16_t* __restrict__ dst, int n) {
  int i = (blockIdx.x * 256 + threadIdx.x) * 8;
  if (i >= n) return;
  F4 a = *reinterpret_cast<const F4*>(src + i);
  F4 b = *reinterpret_cast<const F4*>(src + i + 4);
  U16x8 u;
  u.e[0] = (bf16_t)a.x; u.e[1] = (bf16_t)a.y;
  u.e[2] = (bf16_t)a.z; u.e[3] = (bf16_t)a.w;
  u.e[4] = (bf16_t)b.x; u.e[5] = (bf16_t)b.y;
  u.e[6] = (bf16_t)b.z; u.e[7] = (bf16_t)b.w;
  *reinterpret_cast<Q16*>(dst + i) = u.q;
}

// ---------------------------------------------------------------------------
// Kernel 1: fused QKV projection.
//   grid = B*H*(S/64) blocks of 128 threads (4 waves).
//   Each block: rows [s0,s0+64), head h -> q,k,v tiles [64,64] bf16.
// ---------------------------------------------------------------------------
__global__ __launch_bounds__(128) void qkv_proj_kernel(
    const bf16_t* __restrict__ Qb, const bf16_t* __restrict__ Kb,
    const bf16_t* __restrict__ Vb,
    const bf16_t* __restrict__ Wqb, const bf16_t* __restrict__ Wvb,
    bf16_t* __restrict__ qp, bf16_t* __restrict__ kp, bf16_t* __restrict__ vp) {
  __shared__ bf16_t sQ[64 * 40];   // A tiles [64 rows][32 k], ld 40
  __shared__ bf16_t sK[64 * 40];
  __shared__ bf16_t sV[64 * 40];
  __shared__ bf16_t sWq[64 * 40];  // transposed W: [64 hd][32 k], ld 40
  __shared__ bf16_t sWv[64 * 40];

  const int tid   = threadIdx.x;
  const int w     = tid >> 5;
  const int lane  = tid & 31;
  const int stile = blockIdx.x & 31;
  const int h     = (blockIdx.x >> 5) & 15;
  const int b     = blockIdx.x >> 9;
  const int s0    = stile * 64;

  const size_t xbase = (size_t)b * SS * DD + (size_t)s0 * DD;
  const size_t wbase = (size_t)h * DD * HD;

  v8f accQ[4], accK[4], accV[4];
#pragma unroll
  for (int n = 0; n < 4; ++n) { accQ[n] = zero8(); accK[n] = zero8(); accV[n] = zero8(); }

  const int lr = tid >> 1;            // 0..63  (A-tile row)
  const int lc = (tid & 1) * 16;      // 0/16   (A-tile col chunk)
  const int wr = tid >> 2;            // 0..31  (W-tile row = d)
  const int wc = (tid & 3) * 16;      // 0/16/32/48 (W-tile col chunk = hd)

  for (int kk = 0; kk < DD; kk += 32) {
    // ---- A tiles: 16B async copies straight into LDS
    {
      const bf16_t* gq = Qb + xbase + (size_t)lr * DD + kk + lc;
      const bf16_t* gk = Kb + xbase + (size_t)lr * DD + kk + lc;
      const bf16_t* gv = Vb + xbase + (size_t)lr * DD + kk + lc;
      copy16(sQ + lr * 40 + lc,     gq);
      copy16(sQ + lr * 40 + lc + 8, gq + 8);
      copy16(sK + lr * 40 + lc,     gk);
      copy16(sK + lr * 40 + lc + 8, gk + 8);
      copy16(sV + lr * 40 + lc,     gv);
      copy16(sV + lr * 40 + lc + 8, gv + 8);
      if (kk + 32 < DD) {  // hint next K-step -> global_prefetch_b8
        __builtin_prefetch((const void*)(gq + 32), 0, 3);
        __builtin_prefetch((const void*)(gk + 32), 0, 3);
        __builtin_prefetch((const void*)(gv + 32), 0, 3);
      }
    }
    // ---- W tiles: load row (d), scatter transposed into [hd][k] (VGPR path)
    {
      U16x8 a0, a1, b0, b1;
      const Q16* gq = reinterpret_cast<const Q16*>(Wqb + wbase + (size_t)(kk + wr) * HD + wc);
      const Q16* gv = reinterpret_cast<const Q16*>(Wvb + wbase + (size_t)(kk + wr) * HD + wc);
      a0.q = gq[0]; a1.q = gq[1];
      b0.q = gv[0]; b1.q = gv[1];
#pragma unroll
      for (int i = 0; i < 8; ++i) {
        sWq[(wc + i) * 40 + wr]     = a0.e[i];
        sWq[(wc + 8 + i) * 40 + wr] = a1.e[i];
        sWv[(wc + i) * 40 + wr]     = b0.e[i];
        sWv[(wc + 8 + i) * 40 + wr] = b1.e[i];
      }
    }
    tiles_wait();        // ASYNCcnt == 0: async tiles resident in LDS
    __syncthreads();

    v16bf faQ = frag_a(sQ + w * 16 * 40, 40);
    v16bf faK = frag_a(sK + w * 16 * 40, 40);
    v16bf faV = frag_a(sV + w * 16 * 40, 40);
#pragma unroll
    for (int n = 0; n < 4; ++n) {
      v16bf fbq = frag_b(sWq + n * 16 * 40, 40);
      accQ[n] = wmma_bf16(faQ, fbq, accQ[n]);
      accK[n] = wmma_bf16(faK, fbq, accK[n]);   // reference bug: K uses Wq
      v16bf fbv = frag_b(sWv + n * 16 * 40, 40);
      accV[n] = wmma_bf16(faV, fbv, accV[n]);
    }
    __syncthreads();
  }

  // ---- store C tiles (bf16, [B,H,S,HD])
  const int colb = lane & 15;
  const int rowh = (lane >> 4) << 3;
  const size_t obase = ((size_t)(b * HH + h) * SS + s0 + w * 16) * HD;
#pragma unroll
  for (int n = 0; n < 4; ++n) {
#pragma unroll
    for (int r = 0; r < 8; ++r) {
      const size_t o = obase + (size_t)(rowh + r) * HD + n * 16 + colb;
      qp[o] = (bf16_t)accQ[n][r];
      kp[o] = (bf16_t)accK[n][r];
      vp[o] = (bf16_t)accV[n][r];
    }
  }
}

// ---------------------------------------------------------------------------
// Kernel 2: flash attention (online softmax), 64 query rows per block.
//   grid = B*H*(S/64), 128 threads. Key tiles of 32, full S sweep.
// ---------------------------------------------------------------------------
__global__ __launch_bounds__(128) void attn_kernel(
    const bf16_t* __restrict__ qp, const bf16_t* __restrict__ kp,
    const bf16_t* __restrict__ vp, bf16_t* __restrict__ concat) {
  __shared__ bf16_t sQt[64 * 80];       // q tile [64 rows][64 hd], ld 80
  __shared__ bf16_t sKt[32 * 80];       // k tile [32 keys][64 hd], ld 80 (BT for scores)
  __shared__ bf16_t sVt[64 * 40];       // v tile transposed [64 hd][32 keys], ld 40
  __shared__ bf16_t sP[4 * 16 * 40];    // per-wave P staging [16 rows][32 keys], ld 40

  const int tid   = threadIdx.x;
  const int w     = tid >> 5;
  const int lane  = tid & 31;
  const int stile = blockIdx.x & 31;
  const int h     = (blockIdx.x >> 5) & 15;
  const int b     = blockIdx.x >> 9;
  const int s0    = stile * 64;
  const size_t hb = (size_t)(b * HH + h) * SS * HD;

  // ---- q tile once: 32 contiguous bf16 per thread, async into LDS
  {
    const int r = tid >> 1, seg = (tid & 1) * 32;
    const bf16_t* g = qp + hb + (size_t)(s0 + r) * HD + seg;
    bf16_t* d = sQt + r * 80 + seg;
    copy16(d,      g);
    copy16(d + 8,  g + 8);
    copy16(d + 16, g + 16);
    copy16(d + 24, g + 24);
  }

  float mrow[8], lrow[8];
  v8f   O[4];
#pragma unroll
  for (int r = 0; r < 8; ++r) { mrow[r] = -3.0e38f; lrow[r] = 0.f; }
#pragma unroll
  for (int t = 0; t < 4; ++t) O[t] = zero8();

  const int kr = tid >> 2;           // 0..31 key row
  const int kc = (tid & 3) * 16;     // hd chunk

  for (int t0 = 0; t0 < SS; t0 += 32) {
    // ---- k tile: async row-major; v tile: transposed scatter (VGPR path)
    {
      const bf16_t* gk = kp + hb + (size_t)(t0 + kr) * HD + kc;
      copy16(sKt + kr * 80 + kc,     gk);
      copy16(sKt + kr * 80 + kc + 8, gk + 8);
      U16x8 v0, v1;
      const Q16* gv = reinterpret_cast<const Q16*>(vp + hb + (size_t)(t0 + kr) * HD + kc);
      v0.q = gv[0]; v1.q = gv[1];
#pragma unroll
      for (int i = 0; i < 8; ++i) {
        sVt[(kc + i) * 40 + kr]     = v0.e[i];
        sVt[(kc + 8 + i) * 40 + kr] = v1.e[i];
      }
      if (t0 + 32 < SS) {
        __builtin_prefetch((const void*)(kp + hb + (size_t)(t0 + 32 + kr) * HD + kc), 0, 3);
        __builtin_prefetch((const void*)(vp + hb + (size_t)(t0 + 32 + kr) * HD + kc), 0, 3);
      }
    }
    tiles_wait();
    __syncthreads();   // (A) tiles (and first-iter q) visible

    // ---- scores: S[16x32] = q[16x64] * k^T[64x32]
    v8f sc0 = zero8(), sc1 = zero8();
#pragma unroll
    for (int kk = 0; kk < HD; kk += 32) {
      v16bf aq = frag_a(sQt + w * 16 * 80 + kk, 80);
      v16bf b0 = frag_b(sKt + kk, 80);
      v16bf b1 = frag_b(sKt + 16 * 80 + kk, 80);
      sc0 = wmma_bf16(aq, b0, sc0);
      sc1 = wmma_bf16(aq, b1, sc1);
    }

    // ---- online softmax (rows striped per C layout; reduce over 16-lane groups)
    const float scale = 0.125f;  // 1/sqrt(64)
    const int   prow0 = ((lane >> 4) << 3);
#pragma unroll
    for (int r = 0; r < 8; ++r) {
      float a0 = sc0[r] * scale;
      float a1 = sc1[r] * scale;
      float mx = fmaxf(a0, a1);
      mx = fmaxf(mx, __shfl_xor(mx, 1, 16));
      mx = fmaxf(mx, __shfl_xor(mx, 2, 16));
      mx = fmaxf(mx, __shfl_xor(mx, 4, 16));
      mx = fmaxf(mx, __shfl_xor(mx, 8, 16));
      float mn   = fmaxf(mrow[r], mx);
      float corr = __expf(mrow[r] - mn);
      float p0   = __expf(a0 - mn);
      float p1   = __expf(a1 - mn);
      float ps   = p0 + p1;
      ps += __shfl_xor(ps, 1, 16);
      ps += __shfl_xor(ps, 2, 16);
      ps += __shfl_xor(ps, 4, 16);
      ps += __shfl_xor(ps, 8, 16);
      lrow[r] = lrow[r] * corr + ps;
      mrow[r] = mn;
      O[0][r] *= corr; O[1][r] *= corr; O[2][r] *= corr; O[3][r] *= corr;
      const int pr = prow0 + r;
      sP[(w * 16 + pr) * 40 + (lane & 15)]      = (bf16_t)p0;
      sP[(w * 16 + pr) * 40 + 16 + (lane & 15)] = (bf16_t)p1;
    }
    __syncthreads();   // (B) sP visible

    // ---- O += P[16x32] * V[32x64]
    v16bf aP = frag_a(sP + w * 16 * 40, 40);
#pragma unroll
    for (int t = 0; t < 4; ++t) {
      v16bf bv = frag_b(sVt + t * 16 * 40, 40);
      O[t] = wmma_bf16(aP, bv, O[t]);
    }
    __syncthreads();   // (C) all LDS reads done before next-iter stores
  }

  // ---- normalize + store concat [B,S,H*HD] bf16
  const int colb = lane & 15;
  const int rowh = (lane >> 4) << 3;
#pragma unroll
  for (int r = 0; r < 8; ++r) {
    const float inv = 1.0f / lrow[r];
    const int   row = s0 + w * 16 + rowh + r;
#pragma unroll
    for (int t = 0; t < 4; ++t) {
      concat[((size_t)b * SS + row) * DD + h * HD + t * 16 + colb] =
          (bf16_t)(O[t][r] * inv);
    }
  }
}

// ---------------------------------------------------------------------------
// Kernel 3: out = concat[4096,1024] * Wo^T + bo  (fp32 out)
//   B operand BT[d][e] == Wo row-major, so no transpose needed in LDS.
//   grid = (4096/64) * (1024/64) = 1024 blocks, 128 threads.
// ---------------------------------------------------------------------------
__global__ __launch_bounds__(128) void out_proj_kernel(
    const bf16_t* __restrict__ concat, const bf16_t* __restrict__ Wob,
    const float* __restrict__ bo, float* __restrict__ out) {
  __shared__ bf16_t sA[64 * 40];   // concat tile [64 rows][32 k], ld 40
  __shared__ bf16_t sB[64 * 40];   // Wo tile [64 d][32 e], ld 40

  const int tid   = threadIdx.x;
  const int w     = tid >> 5;
  const int lane  = tid & 31;
  const int rtile = blockIdx.x & 63;
  const int ctile = blockIdx.x >> 6;
  const int r0    = rtile * 64;
  const int c0    = ctile * 64;

  v8f acc[4];
#pragma unroll
  for (int n = 0; n < 4; ++n) acc[n] = zero8();

  const int lr = tid >> 1;
  const int lc = (tid & 1) * 16;

  for (int kk = 0; kk < DD; kk += 32) {
    const bf16_t* ga = concat + (size_t)(r0 + lr) * DD + kk + lc;
    const bf16_t* gb = Wob    + (size_t)(c0 + lr) * DD + kk + lc;
    copy16(sA + lr * 40 + lc,     ga);
    copy16(sA + lr * 40 + lc + 8, ga + 8);
    copy16(sB + lr * 40 + lc,     gb);
    copy16(sB + lr * 40 + lc + 8, gb + 8);
    if (kk + 32 < DD) {
      __builtin_prefetch((const void*)(ga + 32), 0, 3);
      __builtin_prefetch((const void*)(gb + 32), 0, 3);
    }
    tiles_wait();
    __syncthreads();

    v16bf fa = frag_a(sA + w * 16 * 40, 40);
#pragma unroll
    for (int n = 0; n < 4; ++n) {
      v16bf fb = frag_b(sB + n * 16 * 40, 40);
      acc[n] = wmma_bf16(fa, fb, acc[n]);
    }
    __syncthreads();
  }

  const int colb = lane & 15;
  const int rowh = (lane >> 4) << 3;
#pragma unroll
  for (int n = 0; n < 4; ++n) {
#pragma unroll
    for (int r = 0; r < 8; ++r) {
      const int row = r0 + w * 16 + rowh + r;
      const int col = c0 + n * 16 + colb;
      out[(size_t)row * DD + col] = acc[n][r] + bo[col];
    }
  }
}

// ---------------------------------------------------------------------------
// Host launch
// ---------------------------------------------------------------------------
extern "C" void kernel_launch(void* const* d_in, const int* in_sizes, int n_in,
                              void* d_out, int out_size, void* d_ws, size_t ws_size,
                              hipStream_t stream) {
  const float* Q  = (const float*)d_in[0];
  const float* K  = (const float*)d_in[1];
  const float* V  = (const float*)d_in[2];
  const float* Wq = (const float*)d_in[3];
  const float* Wv = (const float*)d_in[4];
  const float* Wo = (const float*)d_in[5];
  const float* bo = (const float*)d_in[6];
  float* out = (float*)d_out;

  char* ws = (char*)d_ws;
  const size_t nQKV = (size_t)BB * SS * DD;      // 4,194,304
  const size_t nW   = (size_t)HH * DD * HD;      // 1,048,576
  const size_t nWo  = (size_t)DD * DD;           // 1,048,576
  const size_t nP   = (size_t)BB * HH * SS * HD; // 4,194,304

  bf16_t* Qb  = (bf16_t*)(ws);
  bf16_t* Kb  = Qb + nQKV;
  bf16_t* Vb  = Kb + nQKV;
  bf16_t* Wqb = Vb + nQKV;
  bf16_t* Wvb = Wqb + nW;
  bf16_t* Wob = Wvb + nW;
  bf16_t* qp  = Wob + nWo;
  bf16_t* kp  = qp + nP;
  bf16_t* vp  = kp + nP;
  bf16_t* cc  = vp + nP;   // concat [B,S,D]

  // Kernel 0: conversions (all sizes multiples of 2048)
  convert_bf16_kernel<<<(int)(nQKV / 2048), 256, 0, stream>>>(Q,  Qb,  (int)nQKV);
  convert_bf16_kernel<<<(int)(nQKV / 2048), 256, 0, stream>>>(K,  Kb,  (int)nQKV);
  convert_bf16_kernel<<<(int)(nQKV / 2048), 256, 0, stream>>>(V,  Vb,  (int)nQKV);
  convert_bf16_kernel<<<(int)(nW   / 2048), 256, 0, stream>>>(Wq, Wqb, (int)nW);
  convert_bf16_kernel<<<(int)(nW   / 2048), 256, 0, stream>>>(Wv, Wvb, (int)nW);
  convert_bf16_kernel<<<(int)(nWo  / 2048), 256, 0, stream>>>(Wo, Wob, (int)nWo);

  // Kernel 1: QKV projection  (B*H*(S/64) = 1024 blocks)
  qkv_proj_kernel<<<BB * HH * (SS / 64), 128, 0, stream>>>(Qb, Kb, Vb, Wqb, Wvb, qp, kp, vp);

  // Kernel 2: flash attention (1024 blocks)
  attn_kernel<<<BB * HH * (SS / 64), 128, 0, stream>>>(qp, kp, vp, cc);

  // Kernel 3: output projection ((B*S/64)*(D/64) = 1024 blocks)
  out_proj_kernel<<<(BB * SS / 64) * (DD / 64), 128, 0, stream>>>(cc, Wob, bo, out);
}
